// QATGPT2MLP_65429531787885
// MI455X (gfx1250) — 8-bit matrix operand order repaired and checked by reading, compiled, not run on hardware
//
#include <hip/hip_runtime.h>
#include <cstdint>
#include <cstddef>


typedef int v8i __attribute__((ext_vector_type(8)));

union AFrag { v8i v; long long l[4]; };
union BFrag { v8i v; int4      q[2]; };


__global__ void zero_scales_kernel(unsigned* s) {
    if (threadIdx.x < 16) s[threadIdx.x] = 0u;
}

__global__ __launch_bounds__(256)
void absmax_kernel(const float* __restrict__ in, unsigned* __restrict__ out, size_t n) {
    __shared__ float sm[256];
    float m = 0.0f;
    for (size_t i = (size_t)blockIdx.x * blockDim.x + threadIdx.x; i < n;
         i += (size_t)gridDim.x * blockDim.x)
        m = fmaxf(m, fabsf(in[i]));
    sm[threadIdx.x] = m;
    __syncthreads();
    for (int s2 = 128; s2 > 0; s2 >>= 1) {
        if ((int)threadIdx.x < s2) sm[threadIdx.x] = fmaxf(sm[threadIdx.x], sm[threadIdx.x + s2]);
        __syncthreads();
    }
    if (threadIdx.x == 0) atomicMax(out, __float_as_uint(sm[0]));
}

__global__ __launch_bounds__(256)
void quantize_i8_kernel(const float* __restrict__ in, signed char* __restrict__ out,
                        const float* __restrict__ amax, size_t n) {
    const float s   = fmaxf(amax[0] * (1.0f / 127.0f), 1e-8f);
    const float inv = 1.0f / s;
    for (size_t i = (size_t)blockIdx.x * blockDim.x + threadIdx.x; i < n;
         i += (size_t)gridDim.x * blockDim.x) {
        float q = rintf(in[i] * inv);
        q = fminf(fmaxf(q, -127.0f), 127.0f);
        out[i] = (signed char)(int)q;
    }
}


__global__ __launch_bounds__(256)
void lora_xa_kernel(const float* __restrict__ x, const float* __restrict__ A,
                    float* __restrict__ t, int D) {
    const int m = blockIdx.x;
    const float* xr = x + (size_t)m * D;
    float acc[16] = {};
    for (int d = threadIdx.x; d < D; d += 256) {
        const float xv = xr[d];
        #pragma unroll
        for (int r = 0; r < 16; ++r) acc[r] = fmaf(xv, A[(size_t)r * D + d], acc[r]);
    }
    __shared__ float red[256];
    #pragma unroll 1
    for (int r = 0; r < 16; ++r) {
        red[threadIdx.x] = acc[r];
        __syncthreads();
        for (int s2 = 128; s2 > 0; s2 >>= 1) {
            if ((int)threadIdx.x < s2) red[threadIdx.x] += red[threadIdx.x + s2];
            __syncthreads();
        }
        if (threadIdx.x == 0) t[(size_t)m * 16 + r] = red[0];
        __syncthreads();
    }
}


template <bool DO_GELU>
__global__ __launch_bounds__(256)
void gemm_i8_wmma_kernel(const signed char* __restrict__ aq,
                         const signed char* __restrict__ bq,
                         const float* __restrict__ bias,
                         const float* __restrict__ loraT,
                         const float* __restrict__ loraB,
                         const float* __restrict__ amaxA,
                         const float* __restrict__ amaxB,
                         float* __restrict__ outp,
                         int Nt, int K) {
    const int lane = threadIdx.x & 31;
    const int wave = threadIdx.x >> 5;
    const int lr   = lane & 15;
    const int lg   = lane >> 4;
    const int mBase = blockIdx.y * 128 + (wave >> 2) * 64;
    const int nBase = blockIdx.x * 256 + (wave & 3) * 64;

    const float sA  = fmaxf(amaxA[0] * (1.0f / 127.0f), 1e-8f);
    const float sB  = fmaxf(amaxB[0] * (1.0f / 127.0f), 1e-8f);
    const float sAB = sA * sB;

    v8i acc[4][4] = {};

    const signed char* aRow[4];
    #pragma unroll
    for (int i = 0; i < 4; ++i)
        aRow[i] = aq + (size_t)(mBase + i * 16 + lr) * K + 16 * lg;
    const signed char* bRow[4];
    #pragma unroll
    for (int j = 0; j < 4; ++j)
        bRow[j] = bq + (size_t)(nBase + j * 16 + lr) * K + 16 * lg;

    for (int k = 0; k < K; k += 64) {
        AFrag a[4];
        BFrag b[4];
        #pragma unroll
        for (int i = 0; i < 4; ++i) {
            const signed char* p = aRow[i] + k;
            a[i].l[0] = *(const long long*)(p);
            a[i].l[1] = *(const long long*)(p + 8);
            a[i].l[2] = *(const long long*)(p + 32);
            a[i].l[3] = *(const long long*)(p + 40);
        }
        #pragma unroll
        for (int j = 0; j < 4; ++j) {
            const signed char* p = bRow[j] + k;
            b[j].q[0] = *(const int4*)(p);
            b[j].q[1] = *(const int4*)(p + 32);
        }
        #pragma unroll
        for (int i = 0; i < 4; ++i)
            #pragma unroll
            for (int j = 0; j < 4; ++j)
                acc[i][j] = __builtin_amdgcn_wmma_i32_16x16x64_iu8(
                     true, a[i].v,  true, b[j].v,
                    acc[i][j],  false,  false);
    }

    #pragma unroll
    for (int j = 0; j < 4; ++j) {
        const int n  = nBase + j * 16 + lr;
        const float bn = bias[n];
        float Bl[16];
        #pragma unroll
        for (int r = 0; r < 16; ++r) Bl[r] = loraB[(size_t)n * 16 + r];
        #pragma unroll
        for (int i = 0; i < 4; ++i) {
            const int m0 = mBase + i * 16 + 8 * lg;
            #pragma unroll
            for (int v = 0; v < 8; ++v) {
                const int m = m0 + v;
                const float* t = loraT + (size_t)m * 16;
                float lora = 0.0f;
                #pragma unroll
                for (int r = 0; r < 16; ++r) lora = fmaf(t[r], Bl[r], lora);
                float val = (float)acc[i][j][v] * sAB + bn + 2.0f * lora;
                if (DO_GELU)
                    val = 0.5f * val * (1.0f + erff(val * 0.70710678118654752f));
                outp[(size_t)m * Nt + n] = val;
            }
        }
    }
}


extern "C" void kernel_launch(void* const* d_in, const int* in_sizes, int n_in,
                              void* d_out, int out_size, void* d_ws, size_t ws_size,
                              hipStream_t stream) {
    (void)in_sizes; (void)n_in; (void)out_size; (void)ws_size;

    constexpr size_t M = 8192;
    constexpr size_t D = 2048;
    constexpr size_t H = 8192;

    const float* x      = (const float*)d_in[0];
    const float* w_fc   = (const float*)d_in[1];
    const float* b_fc   = (const float*)d_in[2];
    const float* A_fc   = (const float*)d_in[3];
    const float* B_fc   = (const float*)d_in[4];
    const float* w_proj = (const float*)d_in[5];
    const float* b_proj = (const float*)d_in[6];
    const float* A_proj = (const float*)d_in[7];
    const float* B_proj = (const float*)d_in[8];
    float* out = (float*)d_out;

    char* ws = (char*)d_ws;
    size_t off = 0;
    unsigned*    scales = (unsigned*)(ws + off); off += 256;
    signed char* xq     = (signed char*)(ws + off); off += M * D;
    signed char* wfcq   = (signed char*)(ws + off); off += H * D;
    signed char* wprq   = (signed char*)(ws + off); off += D * H;
    signed char* hq     = (signed char*)(ws + off); off += M * H;
    float*       t1     = (float*)(ws + off); off += M * 16 * 4;
    float*       t2     = (float*)(ws + off); off += M * 16 * 4;
    float*       h      = (float*)(ws + off); off += M * H * 4;

    const float* sc_x  = (const float*)(scales + 0);
    const float* sc_w1 = (const float*)(scales + 1);
    const float* sc_w2 = (const float*)(scales + 2);
    const float* sc_h  = (const float*)(scales + 3);

    zero_scales_kernel<<<1, 32, 0, stream>>>(scales);
    absmax_kernel<<<1024, 256, 0, stream>>>(x,      scales + 0, M * D);
    absmax_kernel<<<1024, 256, 0, stream>>>(w_fc,   scales + 1, H * D);
    absmax_kernel<<<1024, 256, 0, stream>>>(w_proj, scales + 2, D * H);

    quantize_i8_kernel<<<2048, 256, 0, stream>>>(x,      xq,   sc_x,  M * D);
    quantize_i8_kernel<<<2048, 256, 0, stream>>>(w_fc,   wfcq, sc_w1, H * D);
    quantize_i8_kernel<<<2048, 256, 0, stream>>>(w_proj, wprq, sc_w2, D * H);

    lora_xa_kernel<<<(unsigned)M, 256, 0, stream>>>(x, A_fc, t1, (int)D);

    gemm_i8_wmma_kernel<true><<<dim3(H / 256, M / 128), 256, 0, stream>>>(
        xq, wfcq, b_fc, t1, B_fc, sc_x, sc_w1, h, (int)H, (int)D);

    absmax_kernel<<<2048, 256, 0, stream>>>(h, scales + 3, M * H);
    quantize_i8_kernel<<<4096, 256, 0, stream>>>(h, hq, sc_h, M * H);
    lora_xa_kernel<<<(unsigned)M, 256, 0, stream>>>(h, A_proj, t2, (int)H);

    gemm_i8_wmma_kernel<false><<<dim3(D / 256, M / 128), 256, 0, stream>>>(
        hq, wprq, b_proj, t2, B_proj, sc_h, sc_w2, out, (int)D, (int)H);
}
